// SpikingProtoNet_5669356830962
// MI455X (gfx1250) — matrix operand order repaired and checked by reading, not compiled, not run on hardware
//
#include <hip/hip_runtime.h>
#include <hip/hip_bf16.h>

// ---------------------------------------------------------------------------
// SpikingProtoNet forward on gfx1250 (MI455X), WMMA f16 path.
//   T=20, B=128, THR=0.05, REF=3
//   conv2 and dense run as implicit GEMM on v_wmma_f32_16x16x32_f16 with the
//   LIF membrane state chained through the WMMA C accumulator (conv2).
// ---------------------------------------------------------------------------

typedef __attribute__((ext_vector_type(16))) _Float16 v16h;
typedef __attribute__((ext_vector_type(8)))  _Float16 v8h;
typedef __attribute__((ext_vector_type(8)))  float    v8f;

union F16x16 { v16h v; v8h h8[2]; };

#define TSTEPS 20
#define BATCH  128
#define THR    0.05f
#define REFC   3.0f

// ---- workspace layout (bytes, 256-aligned) --------------------------------
// tcode : int     [128*28*28]                =    401,408
// p1    : f16     [20,128,14,14,32]          = 32,112,640
// z2    : f16     [20,128,14,14,32]          = 32,112,640
// p2    : f16     [20,128,7,7,32]            =  8,028,160
// wt2   : f16     [32][288]                  =     18,432
// wd16  : f16     [64][1568]                 =    200,704
// y3    : float   [2560][64]                 =    655,360
static constexpr size_t OFF_TCODE = 0;
static constexpr size_t OFF_P1    = 401408;
static constexpr size_t OFF_Z2    = OFF_P1 + 32112640;
static constexpr size_t OFF_P2    = OFF_Z2 + 32112640;
static constexpr size_t OFF_WT2   = OFF_P2 + 8028160;
static constexpr size_t OFF_WD16  = OFF_WT2 + 18432;
static constexpr size_t OFF_Y3    = OFF_WD16 + 200704;

// ---------------------------------------------------------------------------
// 1) latency encoding -> per-pixel spike time code (-1 = no spike)
// ---------------------------------------------------------------------------
__global__ void encode_kernel(const float* __restrict__ x, int* __restrict__ tcode, int n)
{
    int i = blockIdx.x * blockDim.x + threadIdx.x;
    if (i >= n) return;
    float xv = x[i];
    if (xv < THR) { tcode[i] = -1; return; }
    float st = fminf(19.0f * (1.0f - xv), 19.0f * (1.0f - THR));
    int t = (int)rintf(st);                 // round-to-nearest-even == jnp.round
    t = min(max(t, 0), TSTEPS - 1);
    tcode[i] = t;
}

// ---------------------------------------------------------------------------
// 2) weight repack to fp16
//    W2 (OIHW 32x32x3x3)  -> wt2[oc][(dy*3+dx)*32 + ic]
//    Wd [64][c*49+yx]     -> wd16[o][yx*32 + c]   (matches channel-last flatten)
// ---------------------------------------------------------------------------
__global__ void repack_w2_kernel(const float* __restrict__ W2, _Float16* __restrict__ wt2)
{
    int i = blockIdx.x * blockDim.x + threadIdx.x;
    if (i >= 32 * 288) return;
    int oc = i / 288, k = i % 288;
    int tap = k / 32, ic = k % 32;
    wt2[i] = (_Float16)W2[(oc * 32 + ic) * 9 + tap];
}

__global__ void repack_wd_kernel(const float* __restrict__ Wd, _Float16* __restrict__ wd16)
{
    int i = blockIdx.x * blockDim.x + threadIdx.x;
    if (i >= 64 * 1568) return;
    int o = i / 1568, f = i % 1568;       // f = yx*32 + c
    int yx = f / 32, c = f % 32;
    wd16[i] = (_Float16)Wd[o * 1568 + c * 49 + yx];
}

// ---------------------------------------------------------------------------
// 3) fused conv1(1->32, SAME) + LIF + maxpool2 -> p1 [T,B,14,14,32] (f16)
//    Spike input is one-hot in time, so conv == gather of weights where
//    tcode[neighbor]==t.  One thread = one (b, pooled pixel, oc).
// ---------------------------------------------------------------------------
__global__ __launch_bounds__(256)
void conv1_lif_pool_kernel(const int* __restrict__ tcode,
                           const float* __restrict__ W1,   // [32][1][3][3]
                           _Float16* __restrict__ p1)
{
    int oc = threadIdx.x;                                   // 0..31
    int sp = blockIdx.x * blockDim.y + threadIdx.y;         // 0 .. 128*196-1
    if (sp >= BATCH * 196) return;
    int b  = sp / 196;
    int yx = sp % 196;
    int yp = yx / 14, xp = yx % 14;

    float w[9];
#pragma unroll
    for (int i = 0; i < 9; ++i) w[i] = W1[oc * 9 + i];

    int codes[4][4];
    const int* tb = tcode + b * 784;
#pragma unroll
    for (int i = 0; i < 4; ++i)
#pragma unroll
        for (int j = 0; j < 4; ++j) {
            int Y = 2 * yp - 1 + i, X = 2 * xp - 1 + j;
            codes[i][j] = (Y >= 0 && Y < 28 && X >= 0 && X < 28) ? tb[Y * 28 + X] : -9;
        }

    float v[4] = {0.f, 0.f, 0.f, 0.f};
    float r[4] = {0.f, 0.f, 0.f, 0.f};
    for (int t = 0; t < TSTEPS; ++t) {
        float zmax = 0.0f;
#pragma unroll
        for (int pi = 0; pi < 2; ++pi)
#pragma unroll
            for (int pj = 0; pj < 2; ++pj) {
                float acc = 0.0f;
#pragma unroll
                for (int dy = 0; dy < 3; ++dy)
#pragma unroll
                    for (int dx = 0; dx < 3; ++dx)
                        acc += (codes[pi + dy][pj + dx] == t) ? w[dy * 3 + dx] : 0.0f;
                int e = pi * 2 + pj;
                float vv = v[e] + acc;
                float z  = (vv > THR && r[e] <= 0.0f) ? 1.0f : 0.0f;
                vv -= z * THR;
                r[e] = (z > 0.0f) ? REFC : fmaxf(r[e] - 1.0f, 0.0f);
                v[e] = vv;
                zmax = fmaxf(zmax, z);
            }
        p1[((size_t)(t * BATCH + b) * 196 + yp * 14 + xp) * 32 + oc] = (_Float16)zmax;
    }
}

// ---------------------------------------------------------------------------
// 4) fused conv2(32->32, SAME 14x14) + LIF  via WMMA implicit GEMM.
//    Wave tile: M=16 output pixels, N=16 out-channels, K=9 taps x 32 ic.
//    Membrane potential v lives in the f32 accumulator fragment and is fed
//    back through the WMMA C operand each time step.
//    waves = 128 images * 13 Mtiles * 2 Ntiles = 3328
// ---------------------------------------------------------------------------
__global__ __launch_bounds__(256)
void conv2_lif_kernel(const _Float16* __restrict__ p1,   // [T,B,14,14,32]
                      const _Float16* __restrict__ wt2,  // [32][288]
                      _Float16* __restrict__ z2)         // [T,B,14,14,32]
{
    const int lane = threadIdx.x & 31;
    const int wv   = blockIdx.x * (blockDim.x >> 5) + (threadIdx.x >> 5);
    const int b     = wv / 26;
    const int rem   = wv % 26;
    const int mtile = rem >> 1;
    const int ntile = rem & 1;

    const int  hsel = (lane >= 16);        // A/B upper-half lane group
    const int  m    = lane & 15;
    const int  p    = mtile * 16 + m;      // output pixel index (M row)
    const bool pval = (p < 196);
    const int  py   = p / 14, px = p % 14;
    const int  nc   = ntile * 16 + m;      // out-channel (B/D column)

    // Preload B fragments for the 9 taps: 16 contiguous K halves per lane.
    F16x16 bfrag[9];
    const _Float16* wrow = wt2 + nc * 288 + (hsel ? 16 : 0);
#pragma unroll
    for (int tap = 0; tap < 9; ++tap)
        bfrag[tap].v = *(const v16h*)(wrow + tap * 32);

    v8f v, r;
#pragma unroll
    for (int e = 0; e < 8; ++e) { v[e] = 0.0f; r[e] = 0.0f; }

    for (int t = 0; t < TSTEPS; ++t) {
        const _Float16* in_t = p1 + (size_t)(t * BATCH + b) * 196 * 32;
        v8f acc = v;                        // v_{t-1} enters through WMMA C
#pragma unroll
        for (int tap = 0; tap < 9; ++tap) {
            const int dy = tap / 3 - 1, dx = tap % 3 - 1;
            const int yy = py + dy, xx = px + dx;
            const bool ok = pval && yy >= 0 && yy < 14 && xx >= 0 && xx < 14;
            F16x16 a;
            if (ok) {
                const _Float16* src = in_t + (yy * 14 + xx) * 32 + (hsel ? 16 : 0);
                a.h8[0] = *(const v8h*)(src);        // K {0..7}  (+16 upper lanes)
                a.h8[1] = *(const v8h*)(src + 8);    // K {8..15} (+16 upper lanes)
            } else {
#pragma unroll
                for (int q = 0; q < 16; ++q) a.v[q] = (_Float16)0.0f;
            }
            acc = __builtin_amdgcn_wmma_f32_16x16x32_f16(
                false, a.v, false, bfrag[tap].v, (short)0, acc, false, false);
        }
        // LIF on the accumulator fragment: threshold, soft reset, refractory.
        _Float16 zs[8];
#pragma unroll
        for (int e = 0; e < 8; ++e) {
            float vv = acc[e];
            float z  = (vv > THR && r[e] <= 0.0f) ? 1.0f : 0.0f;
            vv -= z * THR;
            r[e] = (z > 0.0f) ? REFC : fmaxf(r[e] - 1.0f, 0.0f);
            acc[e] = vv;
            zs[e]  = (_Float16)z;
        }
        v = acc;
        // scatter spikes (D layout: VGPR e -> M = e (+8 for upper lanes))
        _Float16* out_t = z2 + (size_t)(t * BATCH + b) * 196 * 32 + nc;
#pragma unroll
        for (int e = 0; e < 8; ++e) {
            int pix = mtile * 16 + e + (hsel ? 8 : 0);
            if (pix < 196) out_t[(size_t)pix * 32] = zs[e];
        }
    }
}

// ---------------------------------------------------------------------------
// 5) maxpool2 on spikes: z2 [T,B,14,14,32] -> p2 [T,B,7,7,32]
// ---------------------------------------------------------------------------
__global__ void pool2_kernel(const _Float16* __restrict__ z2, _Float16* __restrict__ p2)
{
    int i = blockIdx.x * blockDim.x + threadIdx.x;   // 2560*49*32
    if (i >= TSTEPS * BATCH * 49 * 32) return;
    int c = i % 32;
    int rest = i / 32;
    int xx = rest % 7; rest /= 7;
    int yy = rest % 7; rest /= 7;
    int tb = rest;
    const _Float16* base = z2 + (size_t)tb * 196 * 32 + c;
    float mx = 0.0f;
#pragma unroll
    for (int a = 0; a < 2; ++a)
#pragma unroll
        for (int bb = 0; bb < 2; ++bb)
            mx = fmaxf(mx, (float)base[((2 * yy + a) * 14 + 2 * xx + bb) * 32]);
    p2[i] = (_Float16)mx;
}

// ---------------------------------------------------------------------------
// 6) dense GEMM via WMMA: y3[2560][64] = p2flat[2560][1568] x wd16^T
//    waves = 160 Mtiles * 4 Ntiles = 640
// ---------------------------------------------------------------------------
__global__ __launch_bounds__(256)
void dense_kernel(const _Float16* __restrict__ h,    // [2560][1568]
                  const _Float16* __restrict__ wd,   // [64][1568]
                  float* __restrict__ y3)            // [2560][64]
{
    const int lane  = threadIdx.x & 31;
    const int wv    = blockIdx.x * (blockDim.x >> 5) + (threadIdx.x >> 5);
    const int mtile = wv >> 2;
    const int ntile = wv & 3;
    const int hsel  = (lane >= 16);
    const int m     = lane & 15;

    const _Float16* arow = h  + (size_t)(mtile * 16 + m) * 1568 + (hsel ? 16 : 0);
    const _Float16* brow = wd + (size_t)(ntile * 16 + m) * 1568 + (hsel ? 16 : 0);

    v8f acc;
#pragma unroll
    for (int e = 0; e < 8; ++e) acc[e] = 0.0f;

    for (int k = 0; k < 1568; k += 32) {
        F16x16 a, bf;
        a.h8[0] = *(const v8h*)(arow + k);
        a.h8[1] = *(const v8h*)(arow + k + 8);
        bf.v    = *(const v16h*)(brow + k);
        acc = __builtin_amdgcn_wmma_f32_16x16x32_f16(
            false, a.v, false, bf.v, (short)0, acc, false, false);
    }
#pragma unroll
    for (int e = 0; e < 8; ++e) {
        int row = mtile * 16 + e + (hsel ? 8 : 0);
        y3[(size_t)row * 64 + ntile * 16 + m] = acc[e];
    }
}

// ---------------------------------------------------------------------------
// 7) output LIF scan: y3 [T*B][64] -> out [B,T,64]
// ---------------------------------------------------------------------------
__global__ void lif_out_kernel(const float* __restrict__ y3, float* __restrict__ out)
{
    int i = blockIdx.x * blockDim.x + threadIdx.x;   // 128*64
    if (i >= BATCH * 64) return;
    int b = i / 64, o = i % 64;
    float v = 0.0f, r = 0.0f;
    for (int t = 0; t < TSTEPS; ++t) {
        v += y3[(size_t)(t * BATCH + b) * 64 + o];
        float z = (v > THR && r <= 0.0f) ? 1.0f : 0.0f;
        v -= z * THR;
        r = (z > 0.0f) ? REFC : fmaxf(r - 1.0f, 0.0f);
        out[(size_t)(b * TSTEPS + t) * 64 + o] = z;
    }
}

// ---------------------------------------------------------------------------
extern "C" void kernel_launch(void* const* d_in, const int* in_sizes, int n_in,
                              void* d_out, int out_size, void* d_ws, size_t ws_size,
                              hipStream_t stream)
{
    const float* x  = (const float*)d_in[0];   // [128,1,28,28]
    const float* W1 = (const float*)d_in[1];   // [32,1,3,3]
    const float* W2 = (const float*)d_in[2];   // [32,32,3,3]
    const float* Wd = (const float*)d_in[3];   // [64,1568]
    float* out = (float*)d_out;                // [128,20,64]

    char* ws = (char*)d_ws;                    // needs ~70.2 MB
    int*      tcode = (int*)     (ws + OFF_TCODE);
    _Float16* p1    = (_Float16*)(ws + OFF_P1);
    _Float16* z2    = (_Float16*)(ws + OFF_Z2);
    _Float16* p2    = (_Float16*)(ws + OFF_P2);
    _Float16* wt2   = (_Float16*)(ws + OFF_WT2);
    _Float16* wd16  = (_Float16*)(ws + OFF_WD16);
    float*    y3    = (float*)   (ws + OFF_Y3);

    // 1) encode
    encode_kernel<<<(BATCH * 784 + 255) / 256, 256, 0, stream>>>(x, tcode, BATCH * 784);
    // 2) weight repacks
    repack_w2_kernel<<<(32 * 288 + 255) / 256, 256, 0, stream>>>(W2, wt2);
    repack_wd_kernel<<<(64 * 1568 + 255) / 256, 256, 0, stream>>>(Wd, wd16);
    // 3) conv1 + LIF + pool  (blockDim = 32 oc x 8 spatial)
    {
        dim3 blk(32, 8);
        int nsp = BATCH * 196;                 // 25088, divisible by 8
        conv1_lif_pool_kernel<<<nsp / 8, blk, 0, stream>>>(tcode, W1, p1);
    }
    // 4) conv2 + LIF (WMMA): 3328 waves, 8 waves/block
    conv2_lif_kernel<<<3328 / 8, 256, 0, stream>>>(p1, wt2, z2);
    // 5) pool
    pool2_kernel<<<(TSTEPS * BATCH * 49 * 32 + 255) / 256, 256, 0, stream>>>(z2, p2);
    // 6) dense (WMMA): 640 waves, 8 waves/block
    dense_kernel<<<640 / 8, 256, 0, stream>>>(p2, wd16, y3);
    // 7) output LIF
    lif_out_kernel<<<(BATCH * 64 + 255) / 256, 256, 0, stream>>>(y3, out);
}
